// SemanticAlignBlock_52587579572247
// MI455X (gfx1250) — matrix operand order repaired and checked by reading, not compiled, not run on hardware
//
#include <hip/hip_runtime.h>
#include <hip/hip_bf16.h>

typedef __attribute__((ext_vector_type(16))) _Float16 v16h;
typedef __attribute__((ext_vector_type(2)))  _Float16 h2;
typedef __attribute__((ext_vector_type(8)))  float    v8f;

#define Bn 8
#define Cn 64
#define Hn 128
#define Wn 128
#define On 64
#define OFFC 18

// ---------------------------------------------------------------------------
// NCHW f32 -> NHWC f32 transpose (per-b [64][16384] -> [16384][64])
// ---------------------------------------------------------------------------
__global__ void __launch_bounds__(256)
k_transpose(const float* __restrict__ src, float* __restrict__ dst) {
    __shared__ float t[32][65];
    const int blk  = blockIdx.x;          // b*512 + pixblock
    const int b    = blk >> 9;
    const int pix0 = (blk & 511) << 5;    // 32 pixels
    const int tid  = threadIdx.x;
    const int lane = tid & 31;
    const int wv   = tid >> 5;
    #pragma unroll
    for (int pass = 0; pass < 8; ++pass) {
        const int c = wv + pass * 8;                          // 0..63
        t[lane][c] = src[((b * Cn + c) << 14) + pix0 + lane]; // coalesced over pixels
    }
    __syncthreads();
    const int c2 = tid & 63;
    const int pr = tid >> 6;
    #pragma unroll
    for (int pass = 0; pass < 8; ++pass) {
        const int p2 = pr + pass * 4;
        dst[(((b << 14) + pix0 + p2) << 6) + c2] = t[p2][c2]; // coalesced over channels
    }
}

// ---------------------------------------------------------------------------
// Pack deform weights wd[64][64][3][3] f32 into WMMA B-fragments (f16):
// wf[tap][chunk][ntile(4)][lane][half], B(32x16): k = (half<8 ? 8*(lane>>4)+half : 16+8*(lane>>4)+half-8), n = lane&15
// ---------------------------------------------------------------------------
__global__ void __launch_bounds__(256)
k_prep_wfrag(const float* __restrict__ wd, _Float16* __restrict__ wf) {
    const int idx = blockIdx.x * 256 + threadIdx.x;
    if (idx >= 9 * 2 * 4 * 32 * 16) return;
    const int half = idx & 15;
    const int lane = (idx >> 4) & 31;
    const int nt   = (idx >> 9) & 3;
    const int cc   = (idx >> 11) & 1;
    const int tap  = idx >> 12;
    const int k  = (half < 8) ? (8 * (lane >> 4) + half) : (16 + 8 * (lane >> 4) + (half - 8));
    const int c  = cc * 32 + k;
    const int o  = nt * 16 + (lane & 15);
    const int ti = tap / 3, tj = tap % 3;
    wf[idx] = (_Float16)wd[((o * Cn + c) * 3 + ti) * 3 + tj];
}

// ---------------------------------------------------------------------------
// Pack offset-conv weights wo[18][64][3][3] into zero-padded B-frags (N=32):
// wfo[tap][chunk][ntile(2)][lane][half]
// ---------------------------------------------------------------------------
__global__ void __launch_bounds__(256)
k_prep_wfrag_off(const float* __restrict__ wo, _Float16* __restrict__ wfo) {
    const int idx = blockIdx.x * 256 + threadIdx.x;
    if (idx >= 9 * 2 * 2 * 32 * 16) return;
    const int half = idx & 15;
    const int lane = (idx >> 4) & 31;
    const int nt   = (idx >> 9) & 1;
    const int cc   = (idx >> 10) & 1;
    const int tap  = idx >> 11;
    const int k  = (half < 8) ? (8 * (lane >> 4) + half) : (16 + 8 * (lane >> 4) + (half - 8));
    const int c  = cc * 32 + k;
    const int o  = nt * 16 + (lane & 15);
    const int ti = tap / 3, tj = tap % 3;
    wfo[idx] = (o < OFFC) ? (_Float16)wo[((o * Cn + c) * 3 + ti) * 3 + tj]
                          : (_Float16)0.f;
}

// ---------------------------------------------------------------------------
// 3x3 conv via WMMA, NHWC(64) -> NHWC(18).  1 wave = 16 pixels x 32 out (18
// used), K = 9 taps x 64 ch.  Same A-fragment pipeline as deform but with a
// single masked/clamped corner read per tap (no interpolation).
// ---------------------------------------------------------------------------
__global__ void __launch_bounds__(128)
k_offconv_wmma(const float* __restrict__ src, const _Float16* __restrict__ wfo,
               const float* __restrict__ bias, float* __restrict__ off) {
    __shared__ _Float16 lA[4][2][32][16];   // [wave][chunk][lane][half]
    const int wv   = threadIdx.x >> 5;
    const int lane = threadIdx.x & 31;
    const int tile = blockIdx.x * 4 + wv;   // 8192 tiles
    const int x0   = (tile & 7) << 4;
    const int y    = (tile >> 3) & 127;
    const int b    = tile >> 10;

    v8f acc[2];
    #pragma unroll
    for (int nt = 0; nt < 2; ++nt)
        acc[nt] = (v8f){0.f, 0.f, 0.f, 0.f, 0.f, 0.f, 0.f, 0.f};

    const int p  = lane >> 1;   // pixel 0..15
    const int cc = lane & 1;    // 32-channel chunk
    const int xg = x0 + p;

    for (int tap = 0; tap < 9; ++tap) {
        const int ti = tap / 3, tj = tap % 3;
        const int yy = y + ti - 1;
        const int xs = xg + tj - 1;
        const float mk = (yy >= 0 && yy < Hn && xs >= 0 && xs < Wn) ? 1.f : 0.f;
        const int yc = min(max(yy, 0), Hn - 1);
        const int xc = min(max(xs, 0), Wn - 1);
        const float4* ps = (const float4*)(src + (((b * Hn + yc) * Wn + xc) << 6) + (cc << 5));
        #pragma unroll
        for (int q = 0; q < 8; ++q) {
            const float4 a = ps[q];
            const float s0 = mk * a.x, s1 = mk * a.y, s2 = mk * a.z, s3 = mk * a.w;
            {
                const int k = 4 * q;
                const int g = (k < 16) ? (k >> 3) : ((k - 16) >> 3);
                const int h = (k < 16) ? (k & 7) : (8 + ((k - 16) & 7));
                *(h2*)&lA[wv][cc][(g << 4) + p][h] = (h2){(_Float16)s0, (_Float16)s1};
            }
            {
                const int k = 4 * q + 2;
                const int g = (k < 16) ? (k >> 3) : ((k - 16) >> 3);
                const int h = (k < 16) ? (k & 7) : (8 + ((k - 16) & 7));
                *(h2*)&lA[wv][cc][(g << 4) + p][h] = (h2){(_Float16)s2, (_Float16)s3};
            }
        }
        #pragma unroll
        for (int cc2 = 0; cc2 < 2; ++cc2) {
            const v16h afrag = *(const v16h*)(&lA[wv][cc2][lane][0]);
            #pragma unroll
            for (int nt = 0; nt < 2; ++nt) {
                const v16h bfrag =
                    *(const v16h*)(wfo + ((((tap * 2 + cc2) * 2 + nt) * 32 + lane) << 4));
                acc[nt] = __builtin_amdgcn_wmma_f32_16x16x32_f16(
                    false, afrag, false, bfrag, (short)0, acc[nt], false, false);
            }
        }
    }

    const int mhi = lane >> 4;
    const int n   = lane & 15;
    #pragma unroll
    for (int nt = 0; nt < 2; ++nt) {
        const int o = nt * 16 + n;
        if (o < OFFC) {
            const float bv = bias[o];
            #pragma unroll
            for (int r = 0; r < 8; ++r) {
                const int m = r + 8 * mhi;
                off[(((b * Hn + y) * Wn + x0 + m) * OFFC) + o] = acc[nt][r] + bv;
            }
        }
    }
}

// ---------------------------------------------------------------------------
// Deformable conv via WMMA.  1 wave = tile of 16 pixels x 64 out-channels.
// ---------------------------------------------------------------------------
__global__ void __launch_bounds__(128)
k_deform(const float* __restrict__ src, const float* __restrict__ off,
         const _Float16* __restrict__ wf, const float* __restrict__ bias,
         float* __restrict__ dst, int relu, int nchw) {
    __shared__ _Float16 lA[4][2][32][16];   // [wave][chunk][lane][half]
    const int wv   = threadIdx.x >> 5;
    const int lane = threadIdx.x & 31;
    const int tile = blockIdx.x * 4 + wv;   // 8192 tiles
    const int x0   = (tile & 7) << 4;
    const int y    = (tile >> 3) & 127;
    const int b    = tile >> 10;

    v8f acc[4];
    #pragma unroll
    for (int nt = 0; nt < 4; ++nt)
        acc[nt] = (v8f){0.f, 0.f, 0.f, 0.f, 0.f, 0.f, 0.f, 0.f};

    const int p  = lane >> 1;
    const int cc = lane & 1;
    const int xg = x0 + p;
    const float* offp = off + (((b * Hn + y) * Wn + xg) * OFFC);

    for (int tap = 0; tap < 9; ++tap) {
        const int ti = tap / 3, tj = tap % 3;
        const float py = (float)(y + ti - 1) + offp[2 * tap];
        const float px = (float)(xg + tj - 1) + offp[2 * tap + 1];
        const float fy = floorf(py), fx = floorf(px);
        const int   yi = (int)fy, xi = (int)fx;
        const float wy1 = py - fy, wx1 = px - fx;
        const float wy0 = 1.f - wy1, wx0 = 1.f - wx1;
        const bool vy0 = (yi >= 0) && (yi < Hn);
        const bool vy1 = (yi >= -1) && (yi < Hn - 1);
        const bool vx0 = (xi >= 0) && (xi < Wn);
        const bool vx1 = (xi >= -1) && (xi < Wn - 1);
        const float m00 = (vy0 && vx0) ? wy0 * wx0 : 0.f;
        const float m01 = (vy0 && vx1) ? wy0 * wx1 : 0.f;
        const float m10 = (vy1 && vx0) ? wy1 * wx0 : 0.f;
        const float m11 = (vy1 && vx1) ? wy1 * wx1 : 0.f;
        const int yc0 = min(max(yi, 0), Hn - 1),     yc1 = min(max(yi + 1, 0), Hn - 1);
        const int xc0 = min(max(xi, 0), Wn - 1),     xc1 = min(max(xi + 1, 0), Wn - 1);
        const int cb  = cc << 5;
        const float4* p00 = (const float4*)(src + (((b * Hn + yc0) * Wn + xc0) << 6) + cb);
        const float4* p01 = (const float4*)(src + (((b * Hn + yc0) * Wn + xc1) << 6) + cb);
        const float4* p10 = (const float4*)(src + (((b * Hn + yc1) * Wn + xc0) << 6) + cb);
        const float4* p11 = (const float4*)(src + (((b * Hn + yc1) * Wn + xc1) << 6) + cb);
        #pragma unroll
        for (int q = 0; q < 8; ++q) {
            const float4 a = p00[q], bq = p01[q], cq = p10[q], dq = p11[q];
            const float s0 = m00 * a.x + m01 * bq.x + m10 * cq.x + m11 * dq.x;
            const float s1 = m00 * a.y + m01 * bq.y + m10 * cq.y + m11 * dq.y;
            const float s2 = m00 * a.z + m01 * bq.z + m10 * cq.z + m11 * dq.z;
            const float s3 = m00 * a.w + m01 * bq.w + m10 * cq.w + m11 * dq.w;
            {
                const int k = 4 * q;
                const int g = (k < 16) ? (k >> 3) : ((k - 16) >> 3);
                const int h = (k < 16) ? (k & 7) : (8 + ((k - 16) & 7));
                *(h2*)&lA[wv][cc][(g << 4) + p][h] = (h2){(_Float16)s0, (_Float16)s1};
            }
            {
                const int k = 4 * q + 2;
                const int g = (k < 16) ? (k >> 3) : ((k - 16) >> 3);
                const int h = (k < 16) ? (k & 7) : (8 + ((k - 16) & 7));
                *(h2*)&lA[wv][cc][(g << 4) + p][h] = (h2){(_Float16)s2, (_Float16)s3};
            }
        }
        #pragma unroll
        for (int cc2 = 0; cc2 < 2; ++cc2) {
            const v16h afrag = *(const v16h*)(&lA[wv][cc2][lane][0]);
            #pragma unroll
            for (int nt = 0; nt < 4; ++nt) {
                const v16h bfrag =
                    *(const v16h*)(wf + ((((tap * 2 + cc2) * 4 + nt) * 32 + lane) << 4));
                acc[nt] = __builtin_amdgcn_wmma_f32_16x16x32_f16(
                    false, afrag, false, bfrag, (short)0, acc[nt], false, false);
            }
        }
    }

    const int mhi = lane >> 4;
    const int n   = lane & 15;
    #pragma unroll
    for (int nt = 0; nt < 4; ++nt) {
        const int o  = nt * 16 + n;
        const float bv = bias[o];
        #pragma unroll
        for (int r = 0; r < 8; ++r) {
            const int m = r + 8 * mhi;
            float v = acc[nt][r] + bv;
            if (relu) v = fmaxf(v, 0.f);
            if (nchw) dst[((b * On + o) * Hn + y) * Wn + x0 + m] = v;
            else      dst[(((b * Hn + y) * Wn + x0 + m) << 6) + o] = v;
        }
    }
}

// ---------------------------------------------------------------------------
extern "C" void kernel_launch(void* const* d_in, const int* in_sizes, int n_in,
                              void* d_out, int out_size, void* d_ws, size_t ws_size,
                              hipStream_t stream) {
    const float* x   = (const float*)d_in[0];
    const float* wo1 = (const float*)d_in[1];
    const float* bo1 = (const float*)d_in[2];
    const float* wd1 = (const float*)d_in[3];
    const float* bd1 = (const float*)d_in[4];
    const float* wo2 = (const float*)d_in[5];
    const float* bo2 = (const float*)d_in[6];
    const float* wd2 = (const float*)d_in[7];
    const float* bd2 = (const float*)d_in[8];
    float* out = (float*)d_out;

    char* ws = (char*)d_ws;
    const size_t featBytes = (size_t)Bn * Hn * Wn * Cn * sizeof(float);   // 33.5 MB
    const size_t offBytes  = (size_t)Bn * Hn * Wn * OFFC * sizeof(float); // 9.4 MB
    const size_t wfBytes   = (size_t)9 * 2 * 4 * 32 * 16 * sizeof(_Float16);
    const size_t wfoBytes  = (size_t)9 * 2 * 2 * 32 * 16 * sizeof(_Float16);
    size_t pos = 0;
    float*    xT   = (float*)(ws + pos);    pos += featBytes;
    float*    hT   = (float*)(ws + pos);    pos += featBytes;
    float*    off1 = (float*)(ws + pos);    pos += offBytes;
    float*    off2 = (float*)(ws + pos);    pos += offBytes;
    _Float16* wf1  = (_Float16*)(ws + pos); pos += wfBytes;
    _Float16* wf2  = (_Float16*)(ws + pos); pos += wfBytes;
    _Float16* wfo1 = (_Float16*)(ws + pos); pos += wfoBytes;
    _Float16* wfo2 = (_Float16*)(ws + pos); pos += wfoBytes;

    // 1) x NCHW -> NHWC
    k_transpose<<<Bn * 512, 256, 0, stream>>>(x, xT);
    // 2) pack all weights to f16 B-fragments
    k_prep_wfrag<<<144, 256, 0, stream>>>(wd1, wf1);
    k_prep_wfrag<<<144, 256, 0, stream>>>(wd2, wf2);
    k_prep_wfrag_off<<<72, 256, 0, stream>>>(wo1, wfo1);
    k_prep_wfrag_off<<<72, 256, 0, stream>>>(wo2, wfo2);
    // 3) offset1 = conv(x)            (WMMA)
    k_offconv_wmma<<<(Bn * Hn * (Wn / 16)) / 4, 128, 0, stream>>>(xT, wfo1, bo1, off1);
    // 4) h = relu(deform(x, offset1)) (WMMA, NHWC out)
    k_deform<<<(Bn * Hn * (Wn / 16)) / 4, 128, 0, stream>>>(xT, off1, wf1, bd1, hT, 1, 0);
    // 5) offset2 = conv(h)            (WMMA)
    k_offconv_wmma<<<(Bn * Hn * (Wn / 16)) / 4, 128, 0, stream>>>(hT, wfo2, bo2, off2);
    // 6) out = deform(h, offset2)     (WMMA, NCHW out -> d_out)
    k_deform<<<(Bn * Hn * (Wn / 16)) / 4, 128, 0, stream>>>(hT, off2, wf2, bd2, out, 0, 1);

    (void)in_sizes; (void)n_in; (void)out_size; (void)ws_size;
}
